// onestepKernel_79267916415212
// MI455X (gfx1250) — matrix operand order repaired and checked by reading, not compiled, not run on hardware
//
#include <hip/hip_runtime.h>
#include <hip/hip_bf16.h>

// ---------------------------------------------------------------------------
// Fused NNConv for MI455X (gfx1250, wave32, WMMA bf16)
// 32 edges (2 WMMA M-tiles) per block: B-fragments reused across both tiles,
// halving L2 weight traffic vs 16 edges/block.
// ---------------------------------------------------------------------------

#define N_NODES 50000
#define N_EDGES 400000
#define C_CH    32
#define KW      128
#define EB      32            // edges per block (2 M-tiles)

typedef __attribute__((ext_vector_type(16))) __bf16 v16bf;
typedef __attribute__((ext_vector_type(8)))  __bf16 v8bf;
typedef __attribute__((ext_vector_type(8)))  float  v8f;

// Fragment counts: W2 -> frags [0,32), W3 -> [32,64), W4 -> [64,320)
// Each fragment: 32 lanes x 16 bf16 = 512 bf16 = 1 KB.
#define N_FRAGS   320
#define WQ_BF16   (N_FRAGS * 512)
#define WQ_BYTES  (WQ_BF16 * 2)          // 327680

// ---------------------------------------------------------------------------
// helpers
// ---------------------------------------------------------------------------
__device__ __forceinline__ float prelu_f(float x, float a) {
    return x >= 0.f ? x : a * x;
}

// A-fragment (16x32 bf16, M x K) from LDS row-major [16][128] bf16 tile.
// Per ISA: lane(0-15) m=lane, K in {0..7,16..23}; lane(16-31) m=lane-16,
// K in {8..15,24..31}. Two contiguous 16B LDS loads per lane.
__device__ __forceinline__ v16bf load_afrag(const __bf16* h, int nl, int hi, int kt) {
    union { v16bf v; v8bf h2[2]; } u;
    const __bf16* p = h + nl * KW + kt * 32 + hi * 8;
    u.h2[0] = *(const v8bf*)(p);
    u.h2[1] = *(const v8bf*)(p + 16);
    return u.v;
}

// B-fragment (32x16 bf16, K x N) pre-swizzled in global ws: lane-major,
// 16 bf16 per lane -> two contiguous global b128 loads.
__device__ __forceinline__ v16bf load_bfrag(const __bf16* __restrict__ wq, int fid, int lane) {
    union { v16bf v; v8bf h2[2]; } u;
    const v8bf* p = (const v8bf*)(wq + fid * 512 + lane * 16);
    u.h2[0] = p[0];
    u.h2[1] = p[1];
    return u.v;
}

// ---------------------------------------------------------------------------
// kernel: zero aggregation buffers (agg[NN*C] followed by cnt[NN])
// ---------------------------------------------------------------------------
__global__ void zero_kernel(float* __restrict__ p, int n) {
    int i = blockIdx.x * blockDim.x + threadIdx.x;
    if (i < n) p[i] = 0.f;
}

// ---------------------------------------------------------------------------
// kernel: pack W2/W3/W4 (fp32 row-major [K][N]) into bf16 B-fragments.
// B layout: lane(0-15) n=lane, K in {0..7,16..23}; lane(16-31)
// n=lane-16, K in {8..15,24..31}  => k_local = (p<8 ? hi*8+p : 16+hi*8+p-8).
// ---------------------------------------------------------------------------
__global__ void pack_weights(const float* __restrict__ W2,
                             const float* __restrict__ W3,
                             const float* __restrict__ W4,
                             __bf16* __restrict__ wq) {
    const int fid  = blockIdx.x;     // 0..319
    const int lane = threadIdx.x;    // 0..31
    const int nl = lane & 15, hi = lane >> 4;

    const float* W; int nt, kt, Nw;
    if (fid < 32)      { W = W2; nt = fid >> 2;        kt = fid & 3; Nw = 128;  }
    else if (fid < 64) { W = W3; nt = (fid - 32) >> 2; kt = fid & 3; Nw = 128;  }
    else               { W = W4; nt = (fid - 64) >> 2; kt = fid & 3; Nw = 1024; }

    __bf16* dst = wq + fid * 512 + lane * 16;
    const int n = nt * 16 + nl;
#pragma unroll
    for (int p = 0; p < 16; ++p) {
        const int k = kt * 32 + (p < 8 ? hi * 8 + p : 16 + hi * 8 + (p - 8));
        dst[p] = (__bf16)W[k * Nw + n];
    }
}

// ---------------------------------------------------------------------------
// one MLP layer (128 -> 128) over TWO M-tiles: wave wv owns N-tile wv.
// Each B fragment is loaded once and fed to two WMMAs (one per M-tile).
// ---------------------------------------------------------------------------
__device__ __forceinline__ void mlp_layer2(const __bf16* hsrc, __bf16* hdst,
                                           const __bf16* __restrict__ wq, int fragbase,
                                           const float* __restrict__ bvec, float alpha,
                                           int lane, int nl, int hi, int wv) {
    v8f acc0 = {}, acc1 = {};
#pragma unroll
    for (int kt = 0; kt < 4; ++kt) {
        v16bf b  = load_bfrag(wq, fragbase + wv * 4 + kt, lane);
        v16bf a0 = load_afrag(hsrc,            nl, hi, kt);
        v16bf a1 = load_afrag(hsrc + 16 * KW,  nl, hi, kt);
        acc0 = __builtin_amdgcn_wmma_f32_16x16x32_bf16(false, a0, false, b,
                                                       (short)0, acc0, false, false);
        acc1 = __builtin_amdgcn_wmma_f32_16x16x32_bf16(false, a1, false, b,
                                                       (short)0, acc1, false, false);
    }
    const float bv = bvec[wv * 16 + nl];
#pragma unroll
    for (int r = 0; r < 8; ++r) {
        float x0 = prelu_f(acc0[r] + bv, alpha);
        float x1 = prelu_f(acc1[r] + bv, alpha);
        const int off = (r + 8 * hi) * KW + wv * 16 + nl;
        hdst[off]           = (__bf16)x0;
        hdst[off + 16 * KW] = (__bf16)x1;
    }
}

// ---------------------------------------------------------------------------
// main fused edge kernel: 32 edges / block, 256 threads (8 waves)
// ---------------------------------------------------------------------------
__global__ void __launch_bounds__(256)
edge_kernel(const float* __restrict__ edge_attr,
            const int*   __restrict__ eidx,        // [2][E]
            const float* __restrict__ xr,          // [NN][32]
            const float* __restrict__ W1,          // [6][128]
            const float* __restrict__ b1,
            const float* __restrict__ b2,
            const float* __restrict__ b3,
            const float* __restrict__ b4,
            const float* __restrict__ pa1,
            const float* __restrict__ pa2,
            const float* __restrict__ pa3,
            const __bf16* __restrict__ wq,
            float* __restrict__ agg,
            float* __restrict__ cnt) {
    __shared__ __align__(16) __bf16 hA[2 * 16 * KW];   // two M-tiles
    __shared__ __align__(16) __bf16 hB[2 * 16 * KW];
    __shared__ float ea_s[EB * 6];
    __shared__ int   src_s[EB];
    __shared__ int   dst_s[EB];
    __shared__ __align__(16) float xs_s[2 * 16 * 32];  // gathered x[src]
    __shared__ float msg_s[2 * 16 * 33];               // stride 33: bank-safe

    const int tid  = threadIdx.x;
    const int lane = tid & 31;
    const int wv   = tid >> 5;
    const int nl   = lane & 15;
    const int hi   = lane >> 4;
    const int e0   = blockIdx.x * EB;

    // ---- phase A: stage edge features + indices -------------------------
    if (tid < EB * 6) {
        int m = tid / 6, f = tid - m * 6;
        ea_s[m * 6 + f] = edge_attr[(e0 + m) * 6 + f];
    }
    if (tid >= 192 && tid < 224) src_s[tid - 192] = eidx[e0 + (tid - 192)];
    if (tid >= 224)              dst_s[tid - 224] = eidx[N_EDGES + e0 + (tid - 224)];
    __syncthreads();

    // ---- phase B: layer 1 (K=6, VALU) + gather x[src] + zero msg --------
    {
        const int m = tid >> 4, j0 = (tid & 15) * 8;
        const float a1 = pa1[0];
#pragma unroll
        for (int mt = 0; mt < 2; ++mt) {
#pragma unroll
            for (int jj = 0; jj < 8; ++jj) {
                const int j = j0 + jj;
                float v = b1[j];
#pragma unroll
                for (int f = 0; f < 6; ++f)
                    v = fmaf(ea_s[(mt * 16 + m) * 6 + f], W1[f * KW + j], v);
                hA[mt * 16 * KW + m * KW + j] = (__bf16)prelu_f(v, a1);
            }
        }
    }
    for (int i = tid; i < 2 * 512; i += 256) {
        int e = i >> 5, c = i & 31;                 // e in [0,32)
        xs_s[i] = xr[(long long)src_s[e] * C_CH + c];
    }
    for (int i = tid; i < 2 * 16 * 33; i += 256) msg_s[i] = 0.f;
    __syncthreads();

    // ---- layers 2 and 3 (WMMA, B shared across both M-tiles) -----------
    mlp_layer2(hA, hB, wq, 0,  b2, pa2[0], lane, nl, hi, wv);
    __syncthreads();
    mlp_layer2(hB, hA, wq, 32, b3, pa3[0], lane, nl, hi, wv);
    __syncthreads();

    // ---- layer 4 (WMMA, N=1024) fused with per-edge matvec -------------
    // wave wv owns N-tiles nt = wv*8 .. wv*8+7 ; c = nt>>1 is constant per
    // tile and o = (nt&1)*16 + nl, so the matvec collapses into register
    // FMAs against x[src] broadcast-loaded from LDS.
    v16bf A0[4], A1[4];
#pragma unroll
    for (int kt = 0; kt < 4; ++kt) {
        A0[kt] = load_afrag(hA,            nl, hi, kt);
        A1[kt] = load_afrag(hA + 16 * KW,  nl, hi, kt);
    }
    float msg0[16], msg1[16];
#pragma unroll
    for (int i = 0; i < 16; ++i) { msg0[i] = 0.f; msg1[i] = 0.f; }

#pragma unroll
    for (int t = 0; t < 8; ++t) {
        const int nt = wv * 8 + t;
        v8f acc0 = {}, acc1 = {};
#pragma unroll
        for (int kt = 0; kt < 4; ++kt) {
            v16bf b = load_bfrag(wq, 64 + nt * 4 + kt, lane);
            acc0 = __builtin_amdgcn_wmma_f32_16x16x32_bf16(false, A0[kt], false, b,
                                                           (short)0, acc0, false, false);
            acc1 = __builtin_amdgcn_wmma_f32_16x16x32_bf16(false, A1[kt], false, b,
                                                           (short)0, acc1, false, false);
        }
        const float b4v = b4[nt * 16 + nl];
        const int c = nt >> 1, par = t & 1;
#pragma unroll
        for (int r = 0; r < 8; ++r) {
            const int m = r + 8 * hi;
            const float x0 = xs_s[m * 32 + c];            // LDS broadcast
            const float x1 = xs_s[512 + m * 32 + c];
            msg0[par * 8 + r] = fmaf(x0, acc0[r] + b4v, msg0[par * 8 + r]);
            msg1[par * 8 + r] = fmaf(x1, acc1[r] + b4v, msg1[par * 8 + r]);
        }
    }

    // cross-wave reduction of msg into LDS
#pragma unroll
    for (int i = 0; i < 16; ++i) {
        const int par = i >> 3, r = i & 7;
        const int off = (r + 8 * hi) * 33 + par * 16 + nl;
        atomicAdd(&msg_s[off],            msg0[i]);
        atomicAdd(&msg_s[16 * 33 + off],  msg1[i]);
    }
    __syncthreads();

    // ---- scatter-add to global (HW fp32 atomics) ------------------------
    for (int i = tid; i < 2 * 512; i += 256) {
        int e = i >> 5, o = i & 31;                 // e in [0,32)
        int mt = e >> 4, m = e & 15;
        unsafeAtomicAdd(&agg[(long long)dst_s[e] * C_CH + o],
                        msg_s[mt * 16 * 33 + m * 33 + o]);
    }
    if (tid < EB) unsafeAtomicAdd(&cnt[dst_s[tid]], 1.0f);
}

// ---------------------------------------------------------------------------
// finalize: out = prelu(agg/max(cnt,1) + xr@root + bias, a_out)
// ---------------------------------------------------------------------------
__global__ void finalize_kernel(const float* __restrict__ agg,
                                const float* __restrict__ cnt,
                                const float* __restrict__ xr,
                                const float* __restrict__ root,   // [32][32]
                                const float* __restrict__ bias,
                                const float* __restrict__ pa,
                                float* __restrict__ out) {
    const int idx = blockIdx.x * blockDim.x + threadIdx.x;
    if (idx >= N_NODES * C_CH) return;
    const int n = idx >> 5, c = idx & 31;
    float cn = cnt[n];
    cn = cn < 1.f ? 1.f : cn;
    float v = agg[idx] / cn + bias[c];
    const float* x = xr + n * C_CH;
#pragma unroll
    for (int k = 0; k < C_CH; ++k) v = fmaf(x[k], root[k * C_CH + c], v);
    out[idx] = prelu_f(v, pa[0]);
}

// ---------------------------------------------------------------------------
// launch
// ---------------------------------------------------------------------------
extern "C" void kernel_launch(void* const* d_in, const int* in_sizes, int n_in,
                              void* d_out, int out_size, void* d_ws, size_t ws_size,
                              hipStream_t stream) {
    const float* xr        = (const float*)d_in[0];
    const float* edge_attr = (const float*)d_in[1];
    const float* W1        = (const float*)d_in[2];
    const float* b1        = (const float*)d_in[3];
    const float* W2        = (const float*)d_in[4];
    const float* b2        = (const float*)d_in[5];
    const float* W3        = (const float*)d_in[6];
    const float* b3        = (const float*)d_in[7];
    const float* W4        = (const float*)d_in[8];
    const float* b4        = (const float*)d_in[9];
    const float* a1        = (const float*)d_in[10];
    const float* a2        = (const float*)d_in[11];
    const float* a3        = (const float*)d_in[12];
    const float* root      = (const float*)d_in[13];
    const float* bias      = (const float*)d_in[14];
    const float* a_out     = (const float*)d_in[15];
    const int*   eidx      = (const int*)d_in[16];
    float*       out       = (float*)d_out;

    // workspace: [bf16 packed weights 320KB][agg NN*32 f32][cnt NN f32]
    __bf16* wq  = (__bf16*)d_ws;
    float*  agg = (float*)((char*)d_ws + WQ_BYTES);
    float*  cnt = agg + (size_t)N_NODES * C_CH;

    const int nz = N_NODES * C_CH + N_NODES;   // agg + cnt contiguous
    zero_kernel<<<(nz + 1023) / 1024, 1024, 0, stream>>>(agg, nz);

    pack_weights<<<N_FRAGS, 32, 0, stream>>>(W2, W3, W4, wq);

    edge_kernel<<<N_EDGES / EB, 256, 0, stream>>>(
        edge_attr, eidx, xr, W1, b1, b2, b3, b4, a1, a2, a3, wq, agg, cnt);

    finalize_kernel<<<(N_NODES * C_CH + 255) / 256, 256, 0, stream>>>(
        agg, cnt, xr, root, bias, a_out, out);
}
